// SelfAttentionV3_35845797052755
// MI455X (gfx1250) — matrix operand order repaired and checked by reading, not compiled, not run on hardware
//
#include <hip/hip_runtime.h>

#define S_LEN 2048
#define E_DIM 1024
#define B_SZ  4
#define BSROWS (B_SZ * S_LEN)   // 8192
#define E3    (3 * E_DIM)       // 3072
#define LDS_STRIDE 40           // 32 K elements + 8 pad (bf16 units)
#define LDS_TILE (128 * LDS_STRIDE)

typedef __attribute__((ext_vector_type(16))) __bf16 bf16x16;
typedef __attribute__((ext_vector_type(8)))  float  floatx8;

__device__ __forceinline__ unsigned short f2bf(float f) {
  union { float f; unsigned int u; } c; c.f = f;
  unsigned int u = c.u;
  u += 0x7FFFu + ((u >> 16) & 1u);   // round-to-nearest-even
  return (unsigned short)(u >> 16);
}

// ---------------- CDNA5 async global->LDS copy (ASYNCcnt tracked) -----------
// Copies 32 contiguous bytes per lane. offset: is added to BOTH the LDS and
// the global address (CDNA5 ISA), so offset:16 handles the
// second 16B chunk with the same address registers.
__device__ __forceinline__ void async_cp32B(const void* g, void* l) {
  unsigned int loff = (unsigned int)(unsigned long long)l;  // LDS offset = low 32 bits
  asm volatile("global_load_async_to_lds_b128 %0, %1, off\n\t"
               "global_load_async_to_lds_b128 %0, %1, off offset:16"
               :: "v"(loff), "v"(g) : "memory");
}

__device__ __forceinline__ void async_drain() {
  asm volatile("s_wait_asynccnt 0" ::: "memory");
}

// ---------------- LDS staging helpers (256 threads) -------------------------
// As: [128 rows][LDS_STRIDE] bf16, rows = M, inner = K (32 valid)
// Bs: [128 cols][LDS_STRIDE] bf16, outer = N, inner = K (32 valid)

__device__ __forceinline__ void stage_A_f32(unsigned short* As, const float* A,
                                            int rowBase, int lda, int kk, int t) {
  int r = t >> 1, kh = (t & 1) << 4;
  const float4* s = (const float4*)(A + (size_t)(rowBase + r) * lda + kk + kh);
  unsigned short* d = As + r * LDS_STRIDE + kh;
#pragma unroll
  for (int j = 0; j < 4; ++j) {
    float4 q = s[j];
    d[4 * j + 0] = f2bf(q.x); d[4 * j + 1] = f2bf(q.y);
    d[4 * j + 2] = f2bf(q.z); d[4 * j + 3] = f2bf(q.w);
  }
}

// direct bf16 copy -> async DMA to LDS
__device__ __forceinline__ void stage_A_bf16_async(unsigned short* As, const unsigned short* A,
                                                   int rowBase, int lda, int kk, int t) {
  int r = t >> 1, kh = (t & 1) << 4;
  async_cp32B(A + (size_t)(rowBase + r) * lda + kk + kh, As + r * LDS_STRIDE + kh);
}

// B source row-major [Kdim][N] (weights, V): write transposed into Bs[col][k]
__device__ __forceinline__ void stage_B_rowmajor(unsigned short* Bs, const unsigned short* Bsrc,
                                                 int ldb, int kk, int colBase, int t) {
  int k = t >> 3, ch = (t & 7) << 4;
  const uint4* s = (const uint4*)(Bsrc + (size_t)(kk + k) * ldb + colBase + ch);
  uint4 u0 = s[0], u1 = s[1];
  unsigned short tmp[16];
  *(uint4*)(tmp) = u0; *(uint4*)(tmp + 8) = u1;
#pragma unroll
  for (int i = 0; i < 16; ++i) Bs[(ch + i) * LDS_STRIDE + k] = tmp[i];
}

// B = K^T for scores: memory Kmem[key][e] row-major; Bs[col=key][k=e] is a
// direct contiguous copy -> async DMA to LDS
__device__ __forceinline__ void stage_B_keys_async(unsigned short* Bs, const unsigned short* Kmem,
                                                   int colBase, int kk, int t) {
  int key = t >> 1, eh = (t & 1) << 4;
  async_cp32B(Kmem + (size_t)(colBase + key) * E_DIM + kk + eh, Bs + key * LDS_STRIDE + eh);
}

// ---------------- WMMA fragment loads --------------------------------------
// A 16x32 bf16 layout: lane h=l>>4,m=l&15; VGPR i holds K = 2i+16h, +1
__device__ __forceinline__ bf16x16 load_fragA(const unsigned short* As, int row0, int lane) {
  int h = lane >> 4, m = lane & 15;
  const unsigned int* p32 = (const unsigned int*)(As + (row0 + m) * LDS_STRIDE);
  union { unsigned int u[8]; bf16x16 v; } fr;
#pragma unroll
  for (int i = 0; i < 8; ++i) {
    int k = 2 * i + 16 * h;
    fr.u[i] = p32[k >> 1];
  }
  return fr.v;
}

// B 32x16 bf16 layout: lane n=l&15 is column; lanes<16 K=0..15, lanes>=16 K=16..31
__device__ __forceinline__ bf16x16 load_fragB(const unsigned short* Bs, int col0, int lane) {
  int h = lane >> 4, n = lane & 15;
  const unsigned int* p32 = (const unsigned int*)(Bs + (col0 + n) * LDS_STRIDE + 16 * h);
  union { unsigned int u[8]; bf16x16 v; } fr;
#pragma unroll
  for (int i = 0; i < 8; ++i) fr.u[i] = p32[i];
  return fr.v;
}

__device__ __forceinline__ void wmma_tile(const unsigned short* As, const unsigned short* Bs,
                                          int r0, int c0, int lane, floatx8 (&acc)[2][4]) {
  bf16x16 a0 = load_fragA(As, r0, lane);
  bf16x16 a1 = load_fragA(As, r0 + 16, lane);
  bf16x16 b[4];
#pragma unroll
  for (int ni = 0; ni < 4; ++ni) b[ni] = load_fragB(Bs, c0 + 16 * ni, lane);
#pragma unroll
  for (int ni = 0; ni < 4; ++ni) {
    acc[0][ni] = __builtin_amdgcn_wmma_f32_16x16x32_bf16(false, a0, false, b[ni],
                                                         (short)0, acc[0][ni], false, false);
    acc[1][ni] = __builtin_amdgcn_wmma_f32_16x16x32_bf16(false, a1, false, b[ni],
                                                         (short)0, acc[1][ni], false, false);
  }
}

// ---------------- Kernel 1: weight conversion f32 -> bf16 -------------------
__global__ __launch_bounds__(256) void cvt_w_kernel(const float* __restrict__ pw,
                                                    const float* __restrict__ ow,
                                                    unsigned short* __restrict__ wqkv,
                                                    unsigned short* __restrict__ wout) {
  int i = blockIdx.x * 256 + threadIdx.x;
  if (i < E_DIM * E3) wqkv[i] = f2bf(pw[i]);
  if (i < E_DIM * E_DIM) wout[i] = f2bf(ow[i]);
}

// ---------------- Kernel 2: QKV projection ----------------------------------
__global__ __launch_bounds__(256) void gemm_qkv_kernel(const float* __restrict__ X,
                                                       const unsigned short* __restrict__ W,
                                                       const float* __restrict__ bias,
                                                       unsigned short* __restrict__ Q,
                                                       unsigned short* __restrict__ K,
                                                       unsigned short* __restrict__ V) {
  __shared__ unsigned short As[2][LDS_TILE];
  __shared__ unsigned short Bs[2][LDS_TILE];
  const int t = threadIdx.x, lane = t & 31, w = t >> 5;
  const int r0 = (w & 3) * 32, c0 = (w >> 2) * 64;
  const int rowBase = blockIdx.y * 128, colBase = blockIdx.x * 128;
  floatx8 acc[2][4] = {};
  stage_A_f32(As[0], X, rowBase, E_DIM, 0, t);
  stage_B_rowmajor(Bs[0], W, E3, 0, colBase, t);
  __syncthreads();
  int buf = 0;
  for (int kk = 0; kk < E_DIM; kk += 32) {
    if (kk + 32 < E_DIM) {
      stage_A_f32(As[buf ^ 1], X, rowBase, E_DIM, kk + 32, t);
      stage_B_rowmajor(Bs[buf ^ 1], W, E3, kk + 32, colBase, t);
    }
    wmma_tile(As[buf], Bs[buf], r0, c0, lane, acc);
    __syncthreads();
    buf ^= 1;
  }
  const int h = lane >> 4, n = lane & 15;
#pragma unroll
  for (int mi = 0; mi < 2; ++mi)
#pragma unroll
    for (int ni = 0; ni < 4; ++ni)
#pragma unroll
      for (int r = 0; r < 8; ++r) {
        int row = rowBase + r0 + mi * 16 + h * 8 + r;
        int col = colBase + c0 + ni * 16 + n;
        float val = acc[mi][ni][r] + bias[col];
        int which = col >> 10, cc = col & (E_DIM - 1);
        unsigned short* O = (which == 0) ? Q : ((which == 1) ? K : V);
        O[(size_t)row * E_DIM + cc] = f2bf(val);
      }
}

// ---------------- Kernel 3: scores = scale * Q K^T, masked ------------------
__global__ __launch_bounds__(256) void gemm_scores_kernel(const unsigned short* __restrict__ Q,
                                                          const unsigned short* __restrict__ Km,
                                                          const int* __restrict__ mask,
                                                          float* __restrict__ Sc) {
  __shared__ unsigned short As[2][LDS_TILE];
  __shared__ unsigned short Bs[2][LDS_TILE];
  const int t = threadIdx.x, lane = t & 31, w = t >> 5;
  const int r0 = (w & 3) * 32, c0 = (w >> 2) * 64;
  const int rowBase = blockIdx.y * 128, colBase = blockIdx.x * 128;
  const int b = blockIdx.z;
  const unsigned short* Qb = Q + (size_t)b * S_LEN * E_DIM;
  const unsigned short* Kb = Km + (size_t)b * S_LEN * E_DIM;
  const int* Mb = mask + (size_t)b * S_LEN * S_LEN;
  float* Sb = Sc + (size_t)b * S_LEN * S_LEN;
  floatx8 acc[2][4] = {};
  stage_A_bf16_async(As[0], Qb, rowBase, E_DIM, 0, t);
  stage_B_keys_async(Bs[0], Kb, colBase, 0, t);
  async_drain();
  __syncthreads();
  int buf = 0;
  for (int kk = 0; kk < E_DIM; kk += 32) {
    if (kk + 32 < E_DIM) {    // async DMA overlaps with the WMMA work below
      stage_A_bf16_async(As[buf ^ 1], Qb, rowBase, E_DIM, kk + 32, t);
      stage_B_keys_async(Bs[buf ^ 1], Kb, colBase, kk + 32, t);
    }
    wmma_tile(As[buf], Bs[buf], r0, c0, lane, acc);
    async_drain();
    __syncthreads();
    buf ^= 1;
  }
  const int h = lane >> 4, n = lane & 15;
  const float scale = 0.03125f;  // 1/sqrt(1024)
#pragma unroll
  for (int mi = 0; mi < 2; ++mi)
#pragma unroll
    for (int ni = 0; ni < 4; ++ni)
#pragma unroll
      for (int r = 0; r < 8; ++r) {
        int row = rowBase + r0 + mi * 16 + h * 8 + r;
        int col = colBase + c0 + ni * 16 + n;
        float val = acc[mi][ni][r] * scale;
        if (Mb[(size_t)row * S_LEN + col] == 0) val = -1e20f;
        Sb[(size_t)row * S_LEN + col] = val;
      }
}

// ---------------- Kernel 4: row softmax (in place, fp32) --------------------
__global__ __launch_bounds__(256) void softmax_kernel(float* __restrict__ Sc) {
  float* p = Sc + (size_t)blockIdx.x * S_LEN;
  const int t = threadIdx.x;
  __shared__ float red[8];
  float vals[8];
  float m = -3.4e38f;
#pragma unroll
  for (int i = 0; i < 8; ++i) { vals[i] = p[t + 256 * i]; m = fmaxf(m, vals[i]); }
#pragma unroll
  for (int off = 16; off > 0; off >>= 1) m = fmaxf(m, __shfl_xor(m, off, 32));
  if ((t & 31) == 0) red[t >> 5] = m;
  __syncthreads();
  m = red[0];
#pragma unroll
  for (int i = 1; i < 8; ++i) m = fmaxf(m, red[i]);
  __syncthreads();
  float s = 0.0f;
#pragma unroll
  for (int i = 0; i < 8; ++i) { vals[i] = __expf(vals[i] - m); s += vals[i]; }
#pragma unroll
  for (int off = 16; off > 0; off >>= 1) s += __shfl_xor(s, off, 32);
  if ((t & 31) == 0) red[t >> 5] = s;
  __syncthreads();
  s = red[0] + red[1] + red[2] + red[3] + red[4] + red[5] + red[6] + red[7];
  float inv = 1.0f / s;
#pragma unroll
  for (int i = 0; i < 8; ++i) p[t + 256 * i] = vals[i] * inv;
}

// ---------------- Kernel 5: ctx = attn @ V ----------------------------------
__global__ __launch_bounds__(256) void gemm_ctx_kernel(const float* __restrict__ Sc,
                                                       const unsigned short* __restrict__ V,
                                                       unsigned short* __restrict__ Ctx) {
  __shared__ unsigned short As[2][LDS_TILE];
  __shared__ unsigned short Bs[2][LDS_TILE];
  const int t = threadIdx.x, lane = t & 31, w = t >> 5;
  const int r0 = (w & 3) * 32, c0 = (w >> 2) * 64;
  const int rowBase = blockIdx.y * 128, colBase = blockIdx.x * 128;
  const int b = blockIdx.z;
  const float* Ab = Sc + (size_t)b * S_LEN * S_LEN;
  const unsigned short* Vb = V + (size_t)b * S_LEN * E_DIM;
  unsigned short* Cb = Ctx + (size_t)b * S_LEN * E_DIM;
  floatx8 acc[2][4] = {};
  stage_A_f32(As[0], Ab, rowBase, S_LEN, 0, t);
  stage_B_rowmajor(Bs[0], Vb, E_DIM, 0, colBase, t);
  __syncthreads();
  int buf = 0;
  for (int kk = 0; kk < S_LEN; kk += 32) {
    if (kk + 32 < S_LEN) {
      stage_A_f32(As[buf ^ 1], Ab, rowBase, S_LEN, kk + 32, t);
      stage_B_rowmajor(Bs[buf ^ 1], Vb, E_DIM, kk + 32, colBase, t);
    }
    wmma_tile(As[buf], Bs[buf], r0, c0, lane, acc);
    __syncthreads();
    buf ^= 1;
  }
  const int h = lane >> 4, n = lane & 15;
#pragma unroll
  for (int mi = 0; mi < 2; ++mi)
#pragma unroll
    for (int ni = 0; ni < 4; ++ni)
#pragma unroll
      for (int r = 0; r < 8; ++r) {
        int row = rowBase + r0 + mi * 16 + h * 8 + r;
        int col = colBase + c0 + ni * 16 + n;
        Cb[(size_t)row * E_DIM + col] = f2bf(acc[mi][ni][r]);
      }
}

// ---------------- Kernel 6: out = ctx @ out_w + out_b -----------------------
__global__ __launch_bounds__(256) void gemm_out_kernel(const unsigned short* __restrict__ Ctx,
                                                       const unsigned short* __restrict__ W,
                                                       const float* __restrict__ bias,
                                                       float* __restrict__ Out) {
  __shared__ unsigned short As[2][LDS_TILE];
  __shared__ unsigned short Bs[2][LDS_TILE];
  const int t = threadIdx.x, lane = t & 31, w = t >> 5;
  const int r0 = (w & 3) * 32, c0 = (w >> 2) * 64;
  const int rowBase = blockIdx.y * 128, colBase = blockIdx.x * 128;
  floatx8 acc[2][4] = {};
  stage_A_bf16_async(As[0], Ctx, rowBase, E_DIM, 0, t);
  stage_B_rowmajor(Bs[0], W, E_DIM, 0, colBase, t);
  async_drain();
  __syncthreads();
  int buf = 0;
  for (int kk = 0; kk < E_DIM; kk += 32) {
    if (kk + 32 < E_DIM) {
      stage_A_bf16_async(As[buf ^ 1], Ctx, rowBase, E_DIM, kk + 32, t);
      stage_B_rowmajor(Bs[buf ^ 1], W, E_DIM, kk + 32, colBase, t);
    }
    wmma_tile(As[buf], Bs[buf], r0, c0, lane, acc);
    async_drain();
    __syncthreads();
    buf ^= 1;
  }
  const int h = lane >> 4, n = lane & 15;
#pragma unroll
  for (int mi = 0; mi < 2; ++mi)
#pragma unroll
    for (int ni = 0; ni < 4; ++ni)
#pragma unroll
      for (int r = 0; r < 8; ++r) {
        int row = rowBase + r0 + mi * 16 + h * 8 + r;
        int col = colBase + c0 + ni * 16 + n;
        Out[(size_t)row * E_DIM + col] = acc[mi][ni][r] + bias[col];
      }
}

// ---------------- Launch -----------------------------------------------------
extern "C" void kernel_launch(void* const* d_in, const int* in_sizes, int n_in,
                              void* d_out, int out_size, void* d_ws, size_t ws_size,
                              hipStream_t stream) {
  const float* x      = (const float*)d_in[0];
  const int*   mask   = (const int*)d_in[1];
  const float* proj_w = (const float*)d_in[2];
  const float* proj_b = (const float*)d_in[3];
  const float* out_w  = (const float*)d_in[4];
  const float* out_b  = (const float*)d_in[5];
  float* out = (float*)d_out;

  char* ws = (char*)d_ws;
  unsigned short* wqkv = (unsigned short*)(ws);                            // 6 MB
  unsigned short* wout = (unsigned short*)(ws + (size_t)(6)  * (1 << 20)); // 2 MB
  unsigned short* q    = (unsigned short*)(ws + (size_t)(8)  * (1 << 20)); // 16 MB
  unsigned short* k    = (unsigned short*)(ws + (size_t)(24) * (1 << 20)); // 16 MB
  unsigned short* v    = (unsigned short*)(ws + (size_t)(40) * (1 << 20)); // 16 MB
  unsigned short* ctx  = (unsigned short*)(ws + (size_t)(56) * (1 << 20)); // 16 MB
  float*          sc   = (float*)(ws + (size_t)(72) * (1 << 20));          // 64 MB

  cvt_w_kernel<<<(E_DIM * E3 + 255) / 256, 256, 0, stream>>>(proj_w, out_w, wqkv, wout);
  gemm_qkv_kernel<<<dim3(E3 / 128, BSROWS / 128), 256, 0, stream>>>(x, wqkv, proj_b, q, k, v);
  gemm_scores_kernel<<<dim3(S_LEN / 128, S_LEN / 128, B_SZ), 256, 0, stream>>>(q, k, mask, sc);
  softmax_kernel<<<B_SZ * S_LEN, 256, 0, stream>>>(sc);
  gemm_ctx_kernel<<<dim3(E_DIM / 128, S_LEN / 128, B_SZ), 256, 0, stream>>>(sc, v, ctx);
  gemm_out_kernel<<<dim3(E_DIM / 128, BSROWS / 128), 256, 0, stream>>>(ctx, wout, out_b, out);
}
